// RelPosMultiHeadAttention_29695403885037
// MI455X (gfx1250) — matrix operand order repaired and checked by reading, not compiled, not run on hardware
//
#include <hip/hip_runtime.h>

#define Bc   2
#define Lc   2048
#define Dc   1024
#define Hc   16
#define DKc  64
#define RELSPAN 4095
#define RELOFF  2047

typedef __attribute__((ext_vector_type(16))) __bf16         v16bf;
typedef __attribute__((ext_vector_type(16))) unsigned short v16us;
typedef __attribute__((ext_vector_type(8)))  unsigned short v8us;
typedef __attribute__((ext_vector_type(8)))  float          v8f;

__device__ __forceinline__ unsigned short f2bf(float f) {
  unsigned int u = __float_as_uint(f);
  u += 0x7fffu + ((u >> 16) & 1u);           // round-to-nearest-even
  return (unsigned short)(u >> 16);
}

__device__ __forceinline__ v8f wmma_bf16(v16bf a, v16bf b, v8f c) {
  return __builtin_amdgcn_wmma_f32_16x16x32_bf16(false, a, false, b, (short)0, c, false, false);
}

// ---- WMMA fragment loaders (contiguous k = half*16 + i in both A and B, not the ISA table order; wave32) ----

// A fragment 16x32 bf16 from row-major src: lane(<16)=row M, VGPR0-3 hold K=half*16+0..7,
// VGPR4-7 hold K=half*16+8..15  -> two contiguous 16B loads per lane.
__device__ __forceinline__ v16bf load_a_frag(const unsigned short* src, int ld, int m0, int k0) {
  int lane = threadIdx.x & 31;
  int half = lane >> 4, m = lane & 15;
  const unsigned short* p = src + (size_t)(m0 + m) * ld + k0;
  v8us lo = *(const v8us*)(p + half * 16);
  v8us hi = *(const v8us*)(p + half * 16 + 8);
  v16us r;
#pragma unroll
  for (int i = 0; i < 8; ++i) { r[i] = lo[i]; r[i + 8] = hi[i]; }
  return __builtin_bit_cast(v16bf, r);
}

// B fragment 32x16 bf16 where logical B[k][n] = src[n0+n][k0+k] (i.e. B = src^T slice).
// lane holds col n=lane&15, K = half*16 + i -> one contiguous 32B load per lane.
__device__ __forceinline__ v16bf load_b_trans(const unsigned short* src, int ld, int n0, int k0) {
  int lane = threadIdx.x & 31;
  int half = lane >> 4, n = lane & 15;
  const unsigned short* p = src + (size_t)(n0 + n) * ld + k0 + half * 16;
  v16us r = *(const v16us*)p;
  return __builtin_bit_cast(v16bf, r);
}

// ---- async DMA global->LDS (CDNA5, ASYNCcnt-tracked) ----
__device__ __forceinline__ void async_g2l_b128(unsigned lds_byte_off, const void* gaddr) {
  unsigned long long ga = (unsigned long long)(size_t)gaddr;
  asm volatile("global_load_async_to_lds_b128 %0, %1, off"
               :: "v"(lds_byte_off), "v"(ga) : "memory");
}
__device__ __forceinline__ void wait_asynccnt0() {
  asm volatile("s_wait_asynccnt 0" ::: "memory");
}

// ---- helper kernels ----

__global__ void cast_bf16_kernel(const float* __restrict__ src, unsigned short* __restrict__ dst, int n) {
  int i = blockIdx.x * blockDim.x + threadIdx.x;
  if (i < n) dst[i] = f2bf(src[i]);
}

// bias table: tab[h][rel+2047] = rel_emb[bucket(rel)][h]   (T5 bidirectional, 32 buckets, maxdist 128)
__global__ void bias_table_kernel(const float* __restrict__ rel_emb, float* __restrict__ tab) {
  int idx = blockIdx.x * blockDim.x + threadIdx.x;
  if (idx >= Hc * RELSPAN) return;
  int h = idx / RELSPAN;
  int rel = idx % RELSPAN - RELOFF;
  const int nb = 16, max_exact = 8;
  int bucket = (rel > 0) ? nb : 0;
  int n = rel < 0 ? -rel : rel;
  int v;
  if (n < max_exact) {
    v = n;
  } else {
    float large = (float)max_exact +
        (__logf((float)n / (float)max_exact) / __logf(128.0f / 8.0f)) * (float)(nb - max_exact);
    int li = (int)large;
    v = li < nb - 1 ? li : nb - 1;
  }
  bucket += v;
  tab[idx] = rel_emb[bucket * Hc + h];
}

// ---- GEMM: out[m,n] = sum_k A[m,k] * W[n,k] + bias[n]  (torch Linear, y = x @ W^T + b)
__global__ __launch_bounds__(128) void gemm_bf16_kernel(
    const unsigned short* __restrict__ A, const unsigned short* __restrict__ W,
    const float* __restrict__ bias, unsigned short* __restrict__ outB,
    float* __restrict__ outF, int M, int N, int Kd) {
  int nt = blockIdx.x * blockDim.y + threadIdx.y;
  int mt = blockIdx.y;
  int m0 = mt * 16, n0 = nt * 16;
  if (m0 >= M || n0 >= N) return;

  v8f acc = {};
  for (int k0 = 0; k0 < Kd; k0 += 32) {
    v16bf a = load_a_frag(A, Kd, m0, k0);
    v16bf b = load_b_trans(W, Kd, n0, k0);
    acc = wmma_bf16(a, b, acc);
  }
  int lane = threadIdx.x & 31;
  int half = lane >> 4, col = lane & 15;
  float bv = bias[n0 + col];
#pragma unroll
  for (int r = 0; r < 8; ++r) {
    size_t o = (size_t)(m0 + r + half * 8) * N + n0 + col;
    float v = acc[r] + bv;
    if (outF) outF[o] = v;
    else      outB[o] = f2bf(v);
  }
}

// ---- fused flash-style attention ----
// grid: (L/16/4, H, B); block (32,4): wave w handles q-tile blockIdx.x*4+w.
// All 4 waves share (b,h) and the k-range, so K/V tiles are staged in LDS once per block:
//   K tile: 32x64 bf16 row-major via async global->LDS DMA (ASYNCcnt)
//   V tile: scattered into B-fragment order so fragment loads are contiguous 32B LDS reads
__global__ __launch_bounds__(128) void attn_kernel(
    const unsigned short* __restrict__ Q, const unsigned short* __restrict__ K,
    const unsigned short* __restrict__ V, const int* __restrict__ mask,
    const float* __restrict__ biastab, unsigned short* __restrict__ outBf) {
  __shared__ __align__(32) unsigned short Klds[32 * 64];     // [row 0..31][col 0..63]
  __shared__ __align__(32) unsigned short VBl[4 * 32 * 16];  // [ntile j][lane][elem i]
  __shared__ __align__(32) unsigned short Plds[4][16 * 32];  // per-wave P tile

  int wave = threadIdx.y;
  int lane = threadIdx.x & 31;
  int tid  = wave * 32 + lane;
  int half = lane >> 4, col = lane & 15;
  int b = blockIdx.z, h = blockIdx.y;
  int q0 = (blockIdx.x * 4 + wave) * 16;
  int hd = h * DKc;
  const int* maskb = mask + (size_t)b * Lc * Lc;
  const float* btab = biastab + (size_t)h * RELSPAN;
  unsigned kldsBase = (unsigned)(size_t)(void*)Klds;  // low 32 bits = LDS byte offset

  // Q A-fragments for the dk=64 contraction (two 32-wide chunks), cached in regs
  v16bf qa0 = load_a_frag(Q, Dc, b * Lc + q0, hd);
  v16bf qa1 = load_a_frag(Q, Dc, b * Lc + q0, hd + 32);

  float mrow[8], lrow[8];
  v8f o0 = {}, o1 = {}, o2 = {}, o3 = {};
#pragma unroll
  for (int r = 0; r < 8; ++r) { mrow[r] = -1e30f; lrow[r] = 0.0f; }

  for (int kt = 0; kt < Lc; kt += 32) {
    // ---- stage K tile: 256 chunks of 16B, 2 async DMA per thread ----
    {
      const unsigned short* ksrc = K + (size_t)(b * Lc + kt) * Dc + hd;
#pragma unroll
      for (int c2 = 0; c2 < 2; ++c2) {
        int c = tid + c2 * 128;
        int row = c >> 3, cb = (c & 7) * 8;
        async_g2l_b128(kldsBase + (unsigned)((row * 64 + cb) * 2),
                       ksrc + (size_t)row * Dc + cb);
      }
    }
    // ---- stage V tile into B-fragment order ----
    {
      const unsigned short* vsrc = V + (size_t)(b * Lc + kt) * Dc + hd;
#pragma unroll
      for (int c2 = 0; c2 < 2; ++c2) {
        int c = tid + c2 * 128;
        int row = c >> 3, cb = (c & 7) * 8;          // tile row, col base (8 cols)
        v8us vv = *(const v8us*)(vsrc + (size_t)row * Dc + cb);
        int hf = row >> 4, i = row & 15;
#pragma unroll
        for (int e = 0; e < 8; ++e) {
          int cc = cb + e;
          int j = cc >> 4, n = cc & 15;
          VBl[(((j * 32 + hf * 16 + n) << 4) + i)] = vv[e];
        }
      }
    }
    wait_asynccnt0();
    __syncthreads();

    // V B-fragments: one contiguous 32B LDS read per lane per 16-wide d tile
    v16bf vb0 = __builtin_bit_cast(v16bf, *(const v16us*)&VBl[(0 * 32 + lane) << 4]);
    v16bf vb1 = __builtin_bit_cast(v16bf, *(const v16us*)&VBl[(1 * 32 + lane) << 4]);
    v16bf vb2 = __builtin_bit_cast(v16bf, *(const v16us*)&VBl[(2 * 32 + lane) << 4]);
    v16bf vb3 = __builtin_bit_cast(v16bf, *(const v16us*)&VBl[(3 * 32 + lane) << 4]);

    v8f P[2];
#pragma unroll
    for (int sub = 0; sub < 2; ++sub) {
      int kc0 = kt + sub * 16;
      // B[d][n] = K[kc0+n][hd+d]: row (sub*16+n) of the row-major K tile, contiguous d
      const unsigned short* kp0 = &Klds[(sub * 16 + col) * 64 + half * 16];
      v16bf kb0 = __builtin_bit_cast(v16bf, *(const v16us*)kp0);
      v16bf kb1 = __builtin_bit_cast(v16bf, *(const v16us*)(kp0 + 32));
      v8f s = {};
      s = wmma_bf16(qa0, kb0, s);
      s = wmma_bf16(qa1, kb1, s);
      int kcol = kc0 + col;
#pragma unroll
      for (int r = 0; r < 8; ++r) {
        int q = q0 + r + half * 8;
        float bias = btab[(kcol - q) + RELOFF];
        int mk = maskb[(size_t)q * Lc + kcol];
        s[r] = (mk == 0) ? -1e9f : (s[r] * 0.125f + bias);  // 1/sqrt(64)
      }
      P[sub] = s;
    }

    // online softmax (row = r + half*8 lives across the 16 lanes of each half)
#pragma unroll
    for (int r = 0; r < 8; ++r) {
      float v = fmaxf(P[0][r], P[1][r]);
      v = fmaxf(v, __shfl_xor(v, 1, 32));
      v = fmaxf(v, __shfl_xor(v, 2, 32));
      v = fmaxf(v, __shfl_xor(v, 4, 32));
      v = fmaxf(v, __shfl_xor(v, 8, 32));
      float nm = fmaxf(mrow[r], v);
      float corr = __expf(mrow[r] - nm);
      mrow[r] = nm;
      float p0 = __expf(P[0][r] - nm);
      float p1 = __expf(P[1][r] - nm);
      P[0][r] = p0; P[1][r] = p1;
      float sum = p0 + p1;
      sum += __shfl_xor(sum, 1, 32);
      sum += __shfl_xor(sum, 2, 32);
      sum += __shfl_xor(sum, 4, 32);
      sum += __shfl_xor(sum, 8, 32);
      lrow[r] = lrow[r] * corr + sum;
      o0[r] *= corr; o1[r] *= corr; o2[r] *= corr; o3[r] *= corr;
    }

    // stage P (f32 C-layout) -> own LDS region, re-load in A-fragment layout
    // (same-wave LDS ops are in-order on CDNA5, no barrier needed in between)
    unsigned short* pl = Plds[wave];
#pragma unroll
    for (int r = 0; r < 8; ++r) {
      pl[(r + half * 8) * 32 + col]      = f2bf(P[0][r]);
      pl[(r + half * 8) * 32 + 16 + col] = f2bf(P[1][r]);
    }
    v16bf pa = load_a_frag(pl, 32, 0, 0);

    o0 = wmma_bf16(pa, vb0, o0);
    o1 = wmma_bf16(pa, vb1, o1);
    o2 = wmma_bf16(pa, vb2, o2);
    o3 = wmma_bf16(pa, vb3, o3);

    __syncthreads();   // all waves done with Klds/VBl before restaging
  }

  // normalize and store attention output (bf16) for the final projection GEMM
#pragma unroll
  for (int r = 0; r < 8; ++r) {
    float inv = 1.0f / lrow[r];
    size_t base = (size_t)(b * Lc + q0 + r + half * 8) * Dc + hd + col;
    outBf[base]      = f2bf(o0[r] * inv);
    outBf[base + 16] = f2bf(o1[r] * inv);
    outBf[base + 32] = f2bf(o2[r] * inv);
    outBf[base + 48] = f2bf(o3[r] * inv);
  }
}

extern "C" void kernel_launch(void* const* d_in, const int* in_sizes, int n_in,
                              void* d_out, int out_size, void* d_ws, size_t ws_size,
                              hipStream_t stream) {
  (void)in_sizes; (void)n_in; (void)out_size; (void)ws_size;

  const float* x      = (const float*)d_in[0];
  const int*   mask   = (const int*)d_in[1];
  const float* Wq_w   = (const float*)d_in[2];
  const float* Wq_b   = (const float*)d_in[3];
  const float* Wk_w   = (const float*)d_in[4];
  const float* Wk_b   = (const float*)d_in[5];
  const float* Wv_w   = (const float*)d_in[6];
  const float* Wv_b   = (const float*)d_in[7];
  const float* Wo_w   = (const float*)d_in[8];
  const float* Wo_b   = (const float*)d_in[9];
  const float* rel_emb = (const float*)d_in[10];
  float* out = (float*)d_out;

  const int ML = Bc * Lc;          // 4096 rows
  size_t off = 0;
  auto alloc = [&](size_t bytes) -> void* {
    void* p = (char*)d_ws + off;
    off += (bytes + 255) & ~(size_t)255;
    return p;
  };
  unsigned short* xbf   = (unsigned short*)alloc((size_t)ML * Dc * 2);
  unsigned short* wqbf  = (unsigned short*)alloc((size_t)Dc * Dc * 2);
  unsigned short* wkbf  = (unsigned short*)alloc((size_t)Dc * Dc * 2);
  unsigned short* wvbf  = (unsigned short*)alloc((size_t)Dc * Dc * 2);
  unsigned short* wobf  = (unsigned short*)alloc((size_t)Dc * Dc * 2);
  unsigned short* Qbf   = (unsigned short*)alloc((size_t)ML * Dc * 2);
  unsigned short* Kbf   = (unsigned short*)alloc((size_t)ML * Dc * 2);
  unsigned short* Vbf   = (unsigned short*)alloc((size_t)ML * Dc * 2);
  unsigned short* Abf   = (unsigned short*)alloc((size_t)ML * Dc * 2);  // attn output
  float*          btab  = (float*)alloc((size_t)Hc * RELSPAN * 4);

  // 1) casts + bias table
  {
    int n = ML * Dc;
    cast_bf16_kernel<<<(n + 255) / 256, 256, 0, stream>>>(x, xbf, n);
    int nw = Dc * Dc;
    cast_bf16_kernel<<<(nw + 255) / 256, 256, 0, stream>>>(Wq_w, wqbf, nw);
    cast_bf16_kernel<<<(nw + 255) / 256, 256, 0, stream>>>(Wk_w, wkbf, nw);
    cast_bf16_kernel<<<(nw + 255) / 256, 256, 0, stream>>>(Wv_w, wvbf, nw);
    cast_bf16_kernel<<<(nw + 255) / 256, 256, 0, stream>>>(Wo_w, wobf, nw);
    int nt = Hc * RELSPAN;
    bias_table_kernel<<<(nt + 255) / 256, 256, 0, stream>>>(rel_emb, btab);
  }

  // 2) Q/K/V projections (bf16 WMMA, f32 accumulate, bf16 outputs)
  {
    dim3 blk(32, 4);
    dim3 grd(Dc / 16 / 4, ML / 16);
    gemm_bf16_kernel<<<grd, blk, 0, stream>>>(xbf, wqbf, Wq_b, Qbf, nullptr, ML, Dc, Dc);
    gemm_bf16_kernel<<<grd, blk, 0, stream>>>(xbf, wkbf, Wk_b, Kbf, nullptr, ML, Dc, Dc);
    gemm_bf16_kernel<<<grd, blk, 0, stream>>>(xbf, wvbf, Wv_b, Vbf, nullptr, ML, Dc, Dc);
  }

  // 3) fused attention (flash-style, never materializes [B,H,L,L] scores)
  {
    dim3 blk(32, 4);
    dim3 grd(Lc / 16 / 4, Hc, Bc);
    attn_kernel<<<grd, blk, 0, stream>>>(Qbf, Kbf, Vbf, mask, btab, Abf);
  }

  // 4) output projection -> f32 d_out
  {
    dim3 blk(32, 4);
    dim3 grd(Dc / 16 / 4, ML / 16);
    gemm_bf16_kernel<<<grd, blk, 0, stream>>>(Abf, wobf, Wo_b, nullptr, out, ML, Dc, Dc);
  }
}
